// SparseDense_62758062129168
// MI455X (gfx1250) — compile-verified, matrix operand order checked by reading, not run on hardware
//
#include <hip/hip_runtime.h>

// ---------------------------------------------------------------------------
// SparseDense forward  y = x @ W + b  on MI455X (gfx1250, wave32, WMMA)
//
// Roofline: 68.7 GFLOP vs 537 MB mandatory HBM traffic -> memory bound
// (~23 us @ 23.3 TB/s) provided the math runs on v_wmma_f32_16x16x32_f16.
// x is binary {0,1} -> exact in f16; W rounds at ~5e-4 rel err; f32 accum.
// Each block covers ALL 512 output columns so x is read from HBM exactly
// once; W (1 MB) is pre-transposed/converted to f16 once and stays L2-hot.
//
// Double-buffered LDS (one barrier per K-step) + CDNA5 async-to-LDS for the
// B tile: global_load_async_to_lds_b128 (ASYNCcnt) + s_wait_asynccnt.
// The builtin exists on this toolchain (probe: round-2 diagnostic) and takes
// v4i32 addrspace(1)* / addrspace(3)* pointers.
// ---------------------------------------------------------------------------

typedef __attribute__((ext_vector_type(16))) _Float16 v16h;
typedef __attribute__((ext_vector_type(8)))  _Float16 h8;
typedef __attribute__((ext_vector_type(4)))  _Float16 h4;
typedef __attribute__((ext_vector_type(8)))  float    v8f;
typedef __attribute__((ext_vector_type(4)))  float    f4;

#define N_ROWS 131072
#define DK     512          // D_IN  (GEMM K)
#define DN     512          // D_OUT (GEMM N)
#define BM     128          // rows per block
#define BK     32           // K per step (matches wmma 16x16x32)
#define KSTEPS (DK / BK)    // 16

// --- CDNA5 async global->LDS path (ASYNCcnt), guarded by toolchain probe ---
#if defined(__has_builtin)
#if __has_builtin(__builtin_amdgcn_global_load_async_to_lds_b128)
#define HAVE_ASYNC_LDS 1
#endif
#endif
#ifndef HAVE_ASYNC_LDS
#define HAVE_ASYNC_LDS 0
#endif

#if HAVE_ASYNC_LDS
// builtin parameter type (from the round-2 diagnostic): pointer to
// int __attribute__((vector_size(16))), global / LDS address spaces.
typedef int v4i __attribute__((vector_size(16)));
typedef __attribute__((address_space(1))) v4i gv4i;
typedef __attribute__((address_space(3))) v4i lv4i;

__device__ __forceinline__ void async_b128(const void* gsrc, void* lds_dst) {
    // per-lane 16B async copy, tracked with ASYNCcnt
    __builtin_amdgcn_global_load_async_to_lds_b128(
        (gv4i*)(unsigned long long)(uintptr_t)gsrc,
        (lv4i*)(unsigned)(uintptr_t)lds_dst,
        /*offset=*/0, /*cpol=*/0);
}

__device__ __forceinline__ void wait_async0() {
#if __has_builtin(__builtin_amdgcn_s_wait_asynccnt)
    __builtin_amdgcn_s_wait_asynccnt(0);
#else
    asm volatile("s_wait_asynccnt 0x0" ::: "memory");
#endif
}
#endif

// ---------------------------------------------------------------------------
// Prep: W[k][n] fp32 -> Wt f16, K-step-blocked + transposed:
//   Wt[ks*DN*BK + n*BK + kk] = (f16) W[(ks*BK+kk)*DN + n]
// so one GEMM K-step's B tile is a single contiguous 32KB block whose linear
// layout equals the LDS layout (n-major, kk contiguous).
// ---------------------------------------------------------------------------
__global__ void snn_prep_wt(const float* __restrict__ W, _Float16* __restrict__ Wt) {
    int idx = blockIdx.x * blockDim.x + threadIdx.x;   // 0 .. 262143
    int ks  = idx >> 14;                                // / (DN*BK)
    int rem = idx & 16383;
    int n   = rem >> 5;
    int kk  = rem & 31;
    Wt[idx] = (_Float16)W[(ks * BK + kk) * DN + n];
}

// Load one 16x32 f16 fragment (A-layout; B symmetric, fed from transposed Wt)
// from LDS: lane L<16 holds row/col L, K = 0..7 & 16..23; lane 16+L holds
// K = 8..15 & 24..31.  base points at &lds[row * BK].
__device__ __forceinline__ v16h load_frag16(const _Float16* base, int half8off) {
    h8 lo = *(const h8*)(base + half8off);
    h8 hi = *(const h8*)(base + 16 + half8off);
    return __builtin_shufflevector(lo, hi, 0, 1, 2, 3, 4, 5, 6, 7,
                                           8, 9, 10, 11, 12, 13, 14, 15);
}

__global__ __launch_bounds__(512)
void snn_gemm_wmma(const float*    __restrict__ A,     // x  [N_ROWS x DK] fp32
                   const _Float16* __restrict__ Wt,    // prepped f16 weights
                   const float*    __restrict__ bias,  // [DN]
                   float*          __restrict__ out) { // y  [N_ROWS x DN]
    __shared__ __align__(16) _Float16 lds_a[2][BM * BK];  // 2 x  8 KB
    __shared__ __align__(16) _Float16 lds_b[2][DN * BK];  // 2 x 32 KB

    const int tid  = threadIdx.x;
    const int wave = tid >> 5;        // 16 waves
    const int lane = tid & 31;
    const int wm   = wave & 1;        // 2 waves along M -> 64 rows each
    const int wn   = wave >> 1;       // 8 waves along N -> 64 cols each
    const int half = lane >> 4;
    const int l15  = lane & 15;
    const int h8off = half * 8;
    const int m0   = blockIdx.x * BM;

    // ---- accumulators: 4x4 tiles of 16x16 f32 (128 VGPRs/lane) ----
    v8f acc[4][4];
#pragma unroll
    for (int mt = 0; mt < 4; ++mt)
#pragma unroll
        for (int nt = 0; nt < 4; ++nt)
#pragma unroll
            for (int j = 0; j < 8; ++j) acc[mt][nt][j] = 0.0f;

    // ---- cooperative-load index precompute ----
    // A tile: 128x32 fp32 = 1024 float4 chunks, 2 per thread.
    int ac_row[2], ac_kq[2];
#pragma unroll
    for (int i = 0; i < 2; ++i) {
        int c = tid + i * 512;
        ac_row[i] = c >> 3;
        ac_kq[i]  = c & 7;
    }

    f4 apre[2];
#if !HAVE_ASYNC_LDS
    h8 bpre[4];
#endif

    // ---- prologue: fill buffer 0 ----
#pragma unroll
    for (int i = 0; i < 2; ++i) {
        const f4* src = (const f4*)(A + (size_t)(m0 + ac_row[i]) * DK + ac_kq[i] * 4);
        apre[i] = __builtin_nontemporal_load(src);   // x streamed once: NT
    }
#if HAVE_ASYNC_LDS
#pragma unroll
    for (int i = 0; i < 4; ++i)
        async_b128(Wt + (size_t)(tid + i * 512) * 8,
                   &lds_b[0][(tid + i * 512) * 8]);
#else
#pragma unroll
    for (int i = 0; i < 4; ++i)
        bpre[i] = *(const h8*)(Wt + (size_t)(tid + i * 512) * 8);
#endif

#pragma unroll
    for (int i = 0; i < 2; ++i) {
        h4 hv;
#pragma unroll
        for (int j = 0; j < 4; ++j) hv[j] = (_Float16)apre[i][j];
        *(h4*)&lds_a[0][ac_row[i] * BK + ac_kq[i] * 4] = hv;
    }
#if HAVE_ASYNC_LDS
    wait_async0();
#else
#pragma unroll
    for (int i = 0; i < 4; ++i)
        *(h8*)&lds_b[0][(tid + i * 512) * 8] = bpre[i];
#endif
    __syncthreads();

    // ---- main K loop: double-buffered, ONE barrier per iteration ----
#pragma unroll 1
    for (int kt = 0; kt < KSTEPS; ++kt) {
        const int cur = kt & 1;
        const int nxt = cur ^ 1;

        // issue next tile's traffic first (overlaps with WMMA below);
        // the target buffer was last read in iteration kt-1 and quiesced
        // by that iteration's barrier
        if (kt + 1 < KSTEPS) {
            const int k0 = (kt + 1) * BK;
#pragma unroll
            for (int i = 0; i < 2; ++i) {
                const f4* src = (const f4*)(A + (size_t)(m0 + ac_row[i]) * DK + k0 + ac_kq[i] * 4);
                apre[i] = __builtin_nontemporal_load(src);
            }
            const _Float16* bsrc = Wt + (size_t)(kt + 1) * (DN * BK);
#if HAVE_ASYNC_LDS
#pragma unroll
            for (int i = 0; i < 4; ++i)
                async_b128(bsrc + (size_t)(tid + i * 512) * 8,
                           &lds_b[nxt][(tid + i * 512) * 8]);
#else
#pragma unroll
            for (int i = 0; i < 4; ++i)
                bpre[i] = *(const h8*)(bsrc + (size_t)(tid + i * 512) * 8);
#endif
        }

        // ---- WMMA compute: 4 A-frags x 4 B-frags = 16 wmma per K-step ----
        v16h afrag[4];
#pragma unroll
        for (int mt = 0; mt < 4; ++mt)
            afrag[mt] = load_frag16(&lds_a[cur][(wm * 64 + mt * 16 + l15) * BK], h8off);

#pragma unroll
        for (int nt = 0; nt < 4; ++nt) {
            v16h bfrag = load_frag16(&lds_b[cur][(wn * 64 + nt * 16 + l15) * BK], h8off);
#pragma unroll
            for (int mt = 0; mt < 4; ++mt) {
                acc[mt][nt] = __builtin_amdgcn_wmma_f32_16x16x32_f16(
                    /*neg_a=*/false, afrag[mt],
                    /*neg_b=*/false, bfrag,
                    /*c_mod=*/(short)0, acc[mt][nt],
                    /*reuse_a=*/false, /*reuse_b=*/false);
            }
        }

        if (kt + 1 < KSTEPS) {
#pragma unroll
            for (int i = 0; i < 2; ++i) {
                h4 hv;
#pragma unroll
                for (int j = 0; j < 4; ++j) hv[j] = (_Float16)apre[i][j];
                *(h4*)&lds_a[nxt][ac_row[i] * BK + ac_kq[i] * 4] = hv;
            }
#if HAVE_ASYNC_LDS
            wait_async0();
#else
#pragma unroll
            for (int i = 0; i < 4; ++i)
                *(h8*)&lds_b[nxt][(tid + i * 512) * 8] = bpre[i];
#endif
            __syncthreads();
        }
    }

    // ---- epilogue: C layout -> lane L<16 holds (M=j, N=L); lane 16+L holds
    //      (M=8+j, N=L).  Add bias, nontemporal store (y never re-read). ----
#pragma unroll
    for (int nt = 0; nt < 4; ++nt) {
        const int col = wn * 64 + nt * 16 + l15;
        const float bv = bias[col];
#pragma unroll
        for (int mt = 0; mt < 4; ++mt) {
            const int mrow = m0 + wm * 64 + mt * 16 + half * 8;
#pragma unroll
            for (int j = 0; j < 8; ++j) {
                __builtin_nontemporal_store(acc[mt][nt][j] + bv,
                                            &out[(size_t)(mrow + j) * DN + col]);
            }
        }
    }
}

extern "C" void kernel_launch(void* const* d_in, const int* in_sizes, int n_in,
                              void* d_out, int out_size, void* d_ws, size_t ws_size,
                              hipStream_t stream) {
    const float* x    = (const float*)d_in[0];   // [131072 x 512] fp32 (binary)
    const float* W    = (const float*)d_in[1];   // [512 x 512] fp32
    const float* bias = (const float*)d_in[2];   // [512] fp32
    float*       out  = (float*)d_out;
    _Float16*    Wt   = (_Float16*)d_ws;         // 512 KB scratch

    // 1) transpose+convert W to f16, K-step-blocked (cheap: 1 MB, L2 hot)
    snn_prep_wt<<<1024, 256, 0, stream>>>(W, Wt);
    // 2) memory-bound WMMA GEMM: 1024 blocks x 512 threads (16 waves)
    snn_gemm_wmma<<<N_ROWS / BM, 512, 0, stream>>>(x, Wt, bias, out);
}
